// FirstOrderLatentSDE_2920577761546
// MI455X (gfx1250) — matrix operand order repaired and checked by reading, not compiled, not run on hardware
//
#include <hip/hip_runtime.h>
#include <hip/hip_bf16.h>
#include <math.h>

// ---------------------------------------------------------------------------
// FirstOrderLatentSDE on gfx1250 (MI455X): bf16 WMMA GEMM pipeline.
// B=1024, T=64, D=256, L=256, C=256, H=1024.
// GEMM: block tile 128x128, 8 waves arranged 4(M) x 2(N); each wave computes
// 32x64 via 8 v_wmma_f32_16x16x32_bf16 per k-step (1.5 b128 loads per WMMA).
// ---------------------------------------------------------------------------

typedef __bf16 bf16_t;
typedef __attribute__((ext_vector_type(16))) __bf16 v16bf;
typedef __attribute__((ext_vector_type(8)))  __bf16 v8bf;
typedef __attribute__((ext_vector_type(8)))  float  v8f;

#define B_ 1024
#define T_ 64
#define D_ 256
#define L_ 256
#define C_ 256
#define H_ 1024
#define CTX_ (C_ + D_)            // 512
#define CAT_ (L_ + C_ + D_)       // 768
#define BETA_ (1.0f / 3.0f)
#define EPS_  1e-07f
#define LOG_2PI_ 1.8378770664093453f

// ---------------------------------------------------------------------------
__device__ __forceinline__ float block_reduce_sum_256(float v) {
    __shared__ float sh[256];
    sh[threadIdx.x] = v;
    __syncthreads();
    #pragma unroll
    for (int s = 128; s > 0; s >>= 1) {
        if ((int)threadIdx.x < s) sh[threadIdx.x] += sh[threadIdx.x + s];
        __syncthreads();
    }
    return sh[0];
}

__device__ __forceinline__ v16bf load_a_frag(const bf16_t* aRow) {
    // A-fragment (16x32 bf16), ISA layout: elems 0..7 : K = half*8 + e
    //                                      elems 8..15: K = 16 + half*8 + e-8
    v8bf lo = *(const v8bf*)(aRow);
    v8bf hi = *(const v8bf*)(aRow + 16);
    v16bf f;
    #pragma unroll
    for (int i = 0; i < 8; ++i) { f[i] = lo[i]; f[i + 8] = hi[i]; }
    return f;
}

__device__ __forceinline__ v16bf load_b_frag(const bf16_t* bRow) {
    // B-fragment (32x16) from WT (N,K): elems 0..7: K = half*8 + e; elems 8..15: K = 16 + half*8 + e-8
    v8bf lo = *(const v8bf*)(bRow);
    v8bf hi = *(const v8bf*)(bRow + 16);
    v16bf f;
    #pragma unroll
    for (int i = 0; i < 8; ++i) { f[i] = lo[i]; f[i + 8] = hi[i]; }
    return f;
}

// ---------------------------------------------------------------------------
// Generic bf16 WMMA GEMM:  Out = act(A @ W + bias)
//   A  : (M, K) bf16, row r at A + r*aRowStride (contiguous K within a row)
//   WT : (N, K) bf16 (weights pre-transposed -> contiguous B fragments)
//   MODE: 0 = f32 out, 1 = bf16 out, 2 = bf16 out + tanh
// ---------------------------------------------------------------------------
template <int MODE>
__global__ __launch_bounds__(256) void gemm_wmma_bf16(
    const bf16_t* __restrict__ A, size_t aRowStride,
    const bf16_t* __restrict__ WT,
    const float* __restrict__ bias,
    void* __restrict__ Out, size_t ldo, int K)
{
    const int lane = threadIdx.x & 31;
    const int wave = threadIdx.x >> 5;
    const int half = lane >> 4;
    const int mn   = lane & 15;
    const int row0 = blockIdx.x * 128 + (wave & 3) * 32;
    const int col0 = blockIdx.y * 128 + (wave >> 2) * 64;

    v8f acc[2][4];
    #pragma unroll
    for (int r = 0; r < 2; ++r)
        #pragma unroll
        for (int j = 0; j < 4; ++j)
            #pragma unroll
            for (int i = 0; i < 8; ++i) acc[r][j][i] = 0.0f;

    const bf16_t* aBase0 = A + (size_t)(row0 + mn) * aRowStride + (size_t)(half * 8);
    const bf16_t* aBase1 = aBase0 + (size_t)16 * aRowStride;

    for (int k0 = 0; k0 < K; k0 += 32) {
        const v16bf a0 = load_a_frag(aBase0 + k0);
        const v16bf a1 = load_a_frag(aBase1 + k0);
        #pragma unroll
        for (int j = 0; j < 4; ++j) {
            const bf16_t* bBase =
                WT + (size_t)(col0 + j * 16 + mn) * (size_t)K + (size_t)k0 + (size_t)(half * 8);
            const v16bf b = load_b_frag(bBase);
            acc[0][j] = __builtin_amdgcn_wmma_f32_16x16x32_bf16(
                false, a0, false, b, (short)0, acc[0][j], false, false);
            acc[1][j] = __builtin_amdgcn_wmma_f32_16x16x32_bf16(
                false, a1, false, b, (short)0, acc[1][j], false, false);
        }
    }

    #pragma unroll
    for (int rt = 0; rt < 2; ++rt)
        #pragma unroll
        for (int j = 0; j < 4; ++j) {
            const int col = col0 + j * 16 + mn;
            const float bb = bias ? bias[col] : 0.0f;
            #pragma unroll
            for (int r = 0; r < 8; ++r) {
                const int row = row0 + rt * 16 + half * 8 + r;
                float v = acc[rt][j][r] + bb;
                if (MODE == 2) v = tanhf(v);
                if (MODE == 0)
                    ((float*)Out)[(size_t)row * ldo + col] = v;
                else
                    ((bf16_t*)Out)[(size_t)row * ldo + col] = (bf16_t)v;
            }
        }
}

// ---------------------------------------------------------------------------
// Fused projection GEMM + Gaussian NLL reduction (same tiling as above):
//   x_hat = zs @ proj_w + proj_b ; accumulate per-element
//   [-0.5*((xs - x_hat)/ns)^2 - log(ns) - 0.5*log(2pi)] * (1/B) into acc.
// ---------------------------------------------------------------------------
__global__ __launch_bounds__(256) void proj_loss_wmma(
    const bf16_t* __restrict__ ZS,      // (B*T, L) bf16
    const bf16_t* __restrict__ WT,      // (D, L) = proj_w^T
    const float* __restrict__ bias,     // (D)
    const float* __restrict__ xs,       // (B*T, D) f32
    const float* __restrict__ noise_std,
    float* __restrict__ acc_logpxs)
{
    const int lane = threadIdx.x & 31;
    const int wave = threadIdx.x >> 5;
    const int half = lane >> 4;
    const int mn   = lane & 15;
    const int row0 = blockIdx.x * 128 + (wave & 3) * 32;
    const int col0 = blockIdx.y * 128 + (wave >> 2) * 64;
    const int K = L_;

    v8f acc[2][4];
    #pragma unroll
    for (int r = 0; r < 2; ++r)
        #pragma unroll
        for (int j = 0; j < 4; ++j)
            #pragma unroll
            for (int i = 0; i < 8; ++i) acc[r][j][i] = 0.0f;

    const bf16_t* aBase0 = ZS + (size_t)(row0 + mn) * (size_t)L_ + (size_t)(half * 8);
    const bf16_t* aBase1 = aBase0 + (size_t)16 * L_;

    for (int k0 = 0; k0 < K; k0 += 32) {
        const v16bf a0 = load_a_frag(aBase0 + k0);
        const v16bf a1 = load_a_frag(aBase1 + k0);
        #pragma unroll
        for (int j = 0; j < 4; ++j) {
            const bf16_t* bBase =
                WT + (size_t)(col0 + j * 16 + mn) * (size_t)K + (size_t)k0 + (size_t)(half * 8);
            const v16bf b = load_b_frag(bBase);
            acc[0][j] = __builtin_amdgcn_wmma_f32_16x16x32_bf16(
                false, a0, false, b, (short)0, acc[0][j], false, false);
            acc[1][j] = __builtin_amdgcn_wmma_f32_16x16x32_bf16(
                false, a1, false, b, (short)0, acc[1][j], false, false);
        }
    }

    const float ns = *noise_std;
    const float inv_ns = 1.0f / ns;
    const float cterm = -logf(ns) - 0.5f * LOG_2PI_;
    float local = 0.0f;

    #pragma unroll
    for (int rt = 0; rt < 2; ++rt)
        #pragma unroll
        for (int j = 0; j < 4; ++j) {
            const int col = col0 + j * 16 + mn;
            const float bb = bias[col];
            #pragma unroll
            for (int r = 0; r < 8; ++r) {
                const int row = row0 + rt * 16 + half * 8 + r;
                const float xh = acc[rt][j][r] + bb;
                const float d = (xs[(size_t)row * D_ + col] - xh) * inv_ns;
                local += -0.5f * d * d + cterm;
            }
        }
    const float tot = block_reduce_sum_256(local);
    if (threadIdx.x == 0) atomicAdd(acc_logpxs, tot * (1.0f / (float)B_));
}

// ---------------------------------------------------------------------------
// Elementwise / setup kernels
// ---------------------------------------------------------------------------
__global__ void init_zero_kernel(float* __restrict__ p, int n) {
    const int i = blockIdx.x * blockDim.x + threadIdx.x;
    if (i < n) p[i] = 0.0f;
}

// xs f32 -> xs_bf16 ; also scatter into ctx_bf16 columns [C, C+D)
__global__ void prep_xs_kernel(const float* __restrict__ xs,
                               bf16_t* __restrict__ xs_b,
                               bf16_t* __restrict__ ctx) {
    const size_t i = (size_t)blockIdx.x * blockDim.x + threadIdx.x;
    const float v = xs[i];
    const bf16_t bv = (bf16_t)v;
    xs_b[i] = bv;
    const size_t r = i >> 8;        // / D_
    const size_t c = i & (D_ - 1);  // % D_
    ctx[r * (size_t)CTX_ + (size_t)C_ + c] = bv;
}

// transpose-convert weight (K,N) f32 -> (N,K) bf16
__global__ void wtrans_kernel(const float* __restrict__ W, bf16_t* __restrict__ WT,
                              int K, int N) {
    const size_t i = (size_t)blockIdx.x * blockDim.x + threadIdx.x;
    if (i >= (size_t)K * (size_t)N) return;
    const size_t k = i / (size_t)N;
    const size_t n = i % (size_t)N;
    WT[n * (size_t)K + k] = (bf16_t)W[i];
}

// z0 = qz0_mean + exp(qz0_logstd)*eps0 ; KL0 reduction ; prime cat for t=1
__global__ __launch_bounds__(256) void z0_kernel(
    const float* __restrict__ q,        // (B, 2L)
    const float* __restrict__ eps0,     // (B, L)
    const float* __restrict__ pz_mean,  // (L)
    const float* __restrict__ pz_logstd,// (L)
    const bf16_t* __restrict__ ctx,     // (B*T, CTX)
    float* __restrict__ z_cur,          // (B, L)
    bf16_t* __restrict__ cat,           // (B, CAT)
    bf16_t* __restrict__ zs,            // (B*T, L)
    float* __restrict__ acc_kl0)
{
    const int b = blockIdx.x;
    const int l = threadIdx.x;
    const float m  = q[(size_t)b * (2 * L_) + l];
    const float lg = q[(size_t)b * (2 * L_) + L_ + l];
    const float z  = m + expf(lg) * eps0[(size_t)b * L_ + l];
    z_cur[(size_t)b * L_ + l] = z;
    zs[((size_t)b * T_) * L_ + l] = (bf16_t)z;  // t = 0 slot
    // prime cat = [bf16(z0), ctx[:, 1, :]] for the first scan step
    bf16_t* crow_out = cat + (size_t)b * CAT_;
    const bf16_t* crow_in = ctx + ((size_t)b * T_ + 1) * (size_t)CTX_;
    crow_out[l]           = (bf16_t)z;
    crow_out[L_ + l]      = crow_in[l];
    crow_out[L_ + C_ + l] = crow_in[C_ + l];

    const float pm = pz_mean[l], pl = pz_logstd[l];
    const float dm = m - pm;
    const float kl = pl - lg + (expf(2.0f * lg) + dm * dm) / (2.0f * expf(2.0f * pl)) - 0.5f;
    const float tot = block_reduce_sum_256(kl);
    if (threadIdx.x == 0) atomicAdd(acc_kl0, tot * (1.0f / (float)B_));
}

// Euler-Maruyama step + KL rate accumulation; also writes cat for step t+1
__global__ __launch_bounds__(256) void sde_update_kernel(
    const float* __restrict__ drift, const float* __restrict__ prior,
    const float* __restrict__ gamma, const float* __restrict__ ts,
    const float* __restrict__ noise, const bf16_t* __restrict__ ctx,
    float* __restrict__ z_cur, bf16_t* __restrict__ cat,
    bf16_t* __restrict__ zs, float* __restrict__ dkl, int t)
{
    const int b = blockIdx.x;
    const int l = threadIdx.x;
    const float g = gamma[l];
    const float diff = sqrtf(2.0f / (BETA_ * g));
    const float sgn = (diff > 0.0f) ? 1.0f : ((diff < 0.0f) ? -1.0f : 0.0f);
    const float den = (fabsf(diff) > EPS_) ? diff : sgn * EPS_;
    const float d = drift[(size_t)b * L_ + l];
    const float p = prior[(size_t)b * L_ + l];
    const float r = (d - p) / den;
    const float dt = ts[t] - ts[t - 1];
    const float e = noise[((size_t)(t - 1) * B_ + b) * (size_t)L_ + l];
    const float zn = z_cur[(size_t)b * L_ + l] + d * dt + diff * sqrtf(dt) * e;
    z_cur[(size_t)b * L_ + l] = zn;
    zs[((size_t)b * T_ + t) * (size_t)L_ + l] = (bf16_t)zn;
    if (t + 1 < T_) {  // prime cat = [bf16(z), ctx[:, t+1, :]] for next step
        bf16_t* crow_out = cat + (size_t)b * CAT_;
        const bf16_t* crow_in = ctx + ((size_t)b * T_ + t + 1) * (size_t)CTX_;
        crow_out[l]           = (bf16_t)zn;
        crow_out[L_ + l]      = crow_in[l];
        crow_out[L_ + C_ + l] = crow_in[C_ + l];
    }
    const float tot = block_reduce_sum_256(r * r);
    if (threadIdx.x == 0) dkl[b] += tot * dt;
}

// out[0] = -log_pxs ; out[1] = mean(dkl) + kl0
__global__ __launch_bounds__(256) void finalize_kernel(
    const float* __restrict__ dkl, const float* __restrict__ acc,
    float* __restrict__ out)
{
    float s = 0.0f;
    for (int i = threadIdx.x; i < B_; i += 256) s += dkl[i];
    const float tot = block_reduce_sum_256(s);
    if (threadIdx.x == 0) {
        out[0] = -acc[0];                               // -log_pxs
        out[1] = tot * (1.0f / (float)B_) + acc[1];     // dkl mean + kl0
    }
}

// ---------------------------------------------------------------------------
// Launch
// ---------------------------------------------------------------------------
extern "C" void kernel_launch(void* const* d_in, const int* in_sizes, int n_in,
                              void* d_out, int out_size, void* d_ws, size_t ws_size,
                              hipStream_t stream) {
    (void)in_sizes; (void)n_in; (void)out_size; (void)ws_size;

    const float* xs        = (const float*)d_in[0];
    const float* ts        = (const float*)d_in[1];
    const float* noise_std = (const float*)d_in[2];
    const float* eps0      = (const float*)d_in[3];
    const float* noise     = (const float*)d_in[4];
    const float* enc_w1    = (const float*)d_in[5];
    const float* enc_b1    = (const float*)d_in[6];
    const float* enc_w2    = (const float*)d_in[7];
    const float* enc_b2    = (const float*)d_in[8];
    const float* qz0_w     = (const float*)d_in[9];
    const float* qz0_b     = (const float*)d_in[10];
    const float* f_w1      = (const float*)d_in[11];
    const float* f_b1      = (const float*)d_in[12];
    const float* f_w2      = (const float*)d_in[13];
    const float* f_b2      = (const float*)d_in[14];
    const float* h_w1      = (const float*)d_in[15];
    const float* h_b1      = (const float*)d_in[16];
    const float* h_w2      = (const float*)d_in[17];
    const float* h_b2      = (const float*)d_in[18];
    const float* proj_w    = (const float*)d_in[19];
    const float* proj_b    = (const float*)d_in[20];
    const float* pz0_mean  = (const float*)d_in[21];
    const float* pz0_logstd= (const float*)d_in[22];
    const float* gamma     = (const float*)d_in[23];
    float* out = (float*)d_out;

    const size_t BT = (size_t)B_ * T_;         // 65536
    const size_t CHUNK = 8192;                 // encoder row chunk

    char* p = (char*)d_ws;
    auto take = [&](size_t bytes) {
        char* r = p;
        p += (bytes + 255) & ~(size_t)255;
        return r;
    };
    bf16_t* xs_b    = (bf16_t*)take(BT * D_ * sizeof(bf16_t));
    bf16_t* ctx     = (bf16_t*)take(BT * CTX_ * sizeof(bf16_t));
    bf16_t* zs      = (bf16_t*)take(BT * L_ * sizeof(bf16_t));
    bf16_t* genc    = (bf16_t*)take(CHUNK * H_ * sizeof(bf16_t));
    bf16_t* enc_w1T = (bf16_t*)take((size_t)D_ * H_ * sizeof(bf16_t));
    bf16_t* enc_w2T = (bf16_t*)take((size_t)H_ * C_ * sizeof(bf16_t));
    bf16_t* qz0_wT  = (bf16_t*)take((size_t)CTX_ * 2 * L_ * sizeof(bf16_t));
    bf16_t* f_w1T   = (bf16_t*)take((size_t)CAT_ * H_ * sizeof(bf16_t));
    bf16_t* f_w2T   = (bf16_t*)take((size_t)H_ * L_ * sizeof(bf16_t));
    bf16_t* h_w1T   = (bf16_t*)take((size_t)L_ * H_ * sizeof(bf16_t));
    bf16_t* h_w2T   = (bf16_t*)take((size_t)H_ * L_ * sizeof(bf16_t));
    bf16_t* proj_wT = (bf16_t*)take((size_t)L_ * D_ * sizeof(bf16_t));
    float*  q       = (float*)take((size_t)B_ * 2 * L_ * sizeof(float));
    float*  z_cur   = (float*)take((size_t)B_ * L_ * sizeof(float));
    bf16_t* cat     = (bf16_t*)take((size_t)B_ * CAT_ * sizeof(bf16_t));
    bf16_t* g1      = (bf16_t*)take((size_t)B_ * H_ * sizeof(bf16_t));
    bf16_t* p1      = (bf16_t*)take((size_t)B_ * H_ * sizeof(bf16_t));
    float*  drift   = (float*)take((size_t)B_ * L_ * sizeof(float));
    float*  prior   = (float*)take((size_t)B_ * L_ * sizeof(float));
    float*  dkl     = (float*)take((size_t)B_ * sizeof(float));
    float*  acc     = (float*)take(2 * sizeof(float));  // [0]=log_pxs, [1]=kl0

    // --- zero accumulators + dkl ---
    init_zero_kernel<<<(B_ + 255) / 256, 256, 0, stream>>>(dkl, B_);
    init_zero_kernel<<<1, 256, 0, stream>>>(acc, 2);

    // --- transpose-convert weights to bf16 (N,K) ---
    auto wt = [&](const float* W, bf16_t* WT, int K, int N) {
        const size_t tot = (size_t)K * N;
        wtrans_kernel<<<(unsigned)((tot + 255) / 256), 256, 0, stream>>>(W, WT, K, N);
    };
    wt(enc_w1, enc_w1T, D_, H_);
    wt(enc_w2, enc_w2T, H_, C_);
    wt(qz0_w,  qz0_wT,  CTX_, 2 * L_);
    wt(f_w1,   f_w1T,   CAT_, H_);
    wt(f_w2,   f_w2T,   H_, L_);
    wt(h_w1,   h_w1T,   L_, H_);
    wt(h_w2,   h_w2T,   H_, L_);
    wt(proj_w, proj_wT, L_, D_);

    // --- xs -> bf16 (+ ctx columns [C, C+D)) ---
    prep_xs_kernel<<<(unsigned)(BT * D_ / 256), 256, 0, stream>>>(xs, xs_b, ctx);

    // --- encoder MLP in row chunks: ctx[:, :C] = tanh(xs@W1+b1)@W2+b2 ---
    for (size_t c = 0; c < BT / CHUNK; ++c) {
        gemm_wmma_bf16<2><<<dim3(CHUNK / 128, H_ / 128), 256, 0, stream>>>(
            xs_b + c * CHUNK * D_, (size_t)D_, enc_w1T, enc_b1, genc, (size_t)H_, D_);
        gemm_wmma_bf16<1><<<dim3(CHUNK / 128, C_ / 128), 256, 0, stream>>>(
            genc, (size_t)H_, enc_w2T, enc_b2,
            ctx + c * CHUNK * CTX_, (size_t)CTX_, H_);
    }

    // --- q = ctx[:,0,:] @ qz0_w + qz0_b  (A rows strided by T*CTX) ---
    gemm_wmma_bf16<0><<<dim3(B_ / 128, (2 * L_) / 128), 256, 0, stream>>>(
        ctx, (size_t)T_ * CTX_, qz0_wT, qz0_b, q, (size_t)(2 * L_), CTX_);

    // --- z0 + KL0 (+ prime cat for t=1) ---
    z0_kernel<<<B_, 256, 0, stream>>>(q, eps0, pz0_mean, pz0_logstd, ctx,
                                      z_cur, cat, zs, acc + 1);

    // --- sequential Euler-Maruyama scan: 63 steps ---
    for (int t = 1; t < T_; ++t) {
        // drift_y = tanh(cat @ f_w1 + f_b1) @ f_w2 + f_b2
        gemm_wmma_bf16<2><<<dim3(B_ / 128, H_ / 128), 256, 0, stream>>>(
            cat, (size_t)CAT_, f_w1T, f_b1, g1, (size_t)H_, CAT_);
        gemm_wmma_bf16<0><<<dim3(B_ / 128, L_ / 128), 256, 0, stream>>>(
            g1, (size_t)H_, f_w2T, f_b2, drift, (size_t)L_, H_);
        // prior = tanh(z @ h_w1 + h_b1) @ h_w2 + h_b2  (z = first L cols of cat)
        gemm_wmma_bf16<2><<<dim3(B_ / 128, H_ / 128), 256, 0, stream>>>(
            cat, (size_t)CAT_, h_w1T, h_b1, p1, (size_t)H_, L_);
        gemm_wmma_bf16<0><<<dim3(B_ / 128, L_ / 128), 256, 0, stream>>>(
            p1, (size_t)H_, h_w2T, h_b2, prior, (size_t)L_, H_);
        // z update + KL rate + cat for next step
        sde_update_kernel<<<B_, 256, 0, stream>>>(
            drift, prior, gamma, ts, noise, ctx, z_cur, cat, zs, dkl, t);
    }

    // --- fused projection + Gaussian NLL ---
    proj_loss_wmma<<<dim3((unsigned)(BT / 128), D_ / 128), 256, 0, stream>>>(
        zs, proj_wT, proj_b, xs, noise_std, acc);

    // --- final combine ---
    finalize_kernel<<<1, 256, 0, stream>>>(dkl, acc, out);
}
